// StructuredStateSpaceSequenceModel_55070070670314
// MI455X (gfx1250) — matrix operand order repaired and checked by reading, not compiled, not run on hardware
//
#include <hip/hip_runtime.h>
#include <hip/hip_bf16.h>
#include <cstdint>

// ---------------- model constants ----------------
constexpr int cB = 64, cL = 1024, cH = 512, cN = 32, cNSSM = 64;
constexpr int cLayers = 4, cDFF = 2048, cV = 128, cNS4 = 8;
constexpr size_t NE = (size_t)cB * cH * cL; // elems of one (B,H,L) activation

typedef __attribute__((ext_vector_type(16))) __bf16 v16bf;
typedef __attribute__((ext_vector_type(8)))  float  v8f;
typedef unsigned short ushort_t;

__device__ __forceinline__ ushort_t f2bf(float f) {
    unsigned u = __float_as_uint(f);
    unsigned r = u + 0x7FFFu + ((u >> 16) & 1u);   // RNE
    return (ushort_t)(r >> 16);
}
__device__ __forceinline__ float gelu_t(float x) {
    return 0.5f * x * (1.0f + tanhf(0.7978845608028654f * (x + 0.044715f * x * x * x)));
}
__device__ __forceinline__ float sigm(float x) { return 1.0f / (1.0f + expf(-x)); }

// async copy of one 16B chunk: global (bf16) -> LDS, tracked by ASYNCcnt
__device__ __forceinline__ void async_b128(unsigned lds_off, const void* gp) {
    unsigned long long ga = (unsigned long long)(uintptr_t)gp;
    asm volatile("global_load_async_to_lds_b128 %0, %1, off" :: "v"(lds_off), "v"(ga) : "memory");
}
__device__ __forceinline__ void async_wait0() {
    asm volatile("s_wait_asynccnt 0x0" ::: "memory");
}

// ---------------- f32 -> bf16 bulk convert ----------------
__global__ __launch_bounds__(256) void cvt_kernel(const float* __restrict__ src,
                                                  ushort_t* __restrict__ dst) {
    size_t i = (size_t)blockIdx.x * 256 + threadIdx.x;
    dst[i] = f2bf(src[i]);
}

// ---------------- embedding gather: X[b,h,l] = emb[tok[b,l], h] ----------------
__global__ __launch_bounds__(256) void embed_kernel(const int* __restrict__ tokens,
                                                    const float* __restrict__ emb,
                                                    float* __restrict__ X) {
    size_t i = (size_t)blockIdx.x * 256 + threadIdx.x;
    int l = (int)(i & (cL - 1));
    int h = (int)((i >> 10) & (cH - 1));
    int b = (int)(i >> 19);
    int tok = tokens[((size_t)b << 10) + l];
    X[i] = emb[(size_t)tok * cH + h];
}

// ---------------- SSM coefficients: decay = exp(dt*A), c2 = 2*C*(exp(dt*A)-1)/A ----
__global__ __launch_bounds__(256) void coef_kernel(const float* __restrict__ log_dt,
                                                   const float* __restrict__ A_re,
                                                   const float* __restrict__ A_im,
                                                   const float* __restrict__ C_re,
                                                   const float* __restrict__ C_im,
                                                   float4* __restrict__ coef) {
    int idx = blockIdx.x * 256 + threadIdx.x;       // H*N = 16384
    int h = idx >> 5, n = idx & 31;
    float dt = expf(log_dt[h]);
    float ar = A_re[(h >> 3) * cN + n];
    float ai = A_im[(h >> 3) * cN + n];
    float crv = C_re[h * cN + n], civ = C_im[h * cN + n];
    float xr = dt * ar, xi = dt * ai;
    float er = expf(xr);
    float wr = er * cosf(xi), wi = er * sinf(xi);
    float nr = wr - 1.0f, ni = wi;
    float den = ar * ar + ai * ai;
    float tr = (nr * ar + ni * ai) / den;
    float ti = (ni * ar - nr * ai) / den;
    float dcr = crv * tr - civ * ti;
    float dci = crv * ti + civ * tr;
    coef[idx] = make_float4(wr, wi, 2.0f * dcr, 2.0f * dci);
}

// ---------------- channel LayerNorm -> f32 Z (for scan) and bf16 Zh (for GEMM) ----
__global__ __launch_bounds__(256) void ln_kernel(const float* __restrict__ X,
                                                 const float* __restrict__ w,
                                                 const float* __restrict__ bp,
                                                 float* __restrict__ Z,
                                                 ushort_t* __restrict__ Zh) {
    __shared__ float tile[cH * 9];
    int tid = threadIdx.x;
    int b = blockIdx.y;
    int l0 = blockIdx.x * 8;
    for (int p = 0; p < 4; ++p) {
        int h = p * 128 + (tid >> 1);
        int part = (tid & 1) * 4;
        const float4 v = *(const float4*)&X[((size_t)b * cH + h) * cL + l0 + part];
        tile[h * 9 + part + 0] = v.x; tile[h * 9 + part + 1] = v.y;
        tile[h * 9 + part + 2] = v.z; tile[h * 9 + part + 3] = v.w;
    }
    __syncthreads();
    int wid = tid >> 5, lane = tid & 31;
    float s = 0.f, s2 = 0.f;
    for (int k = 0; k < 16; ++k) {
        float x = tile[(lane + 32 * k) * 9 + wid];
        s += x; s2 += x * x;
    }
    for (int off = 16; off >= 1; off >>= 1) {
        s += __shfl_xor(s, off, 32);
        s2 += __shfl_xor(s2, off, 32);
    }
    float mu = s * (1.0f / cH);
    float var = s2 * (1.0f / cH) - mu * mu;
    float rs = rsqrtf(var + 1e-5f);
    for (int k = 0; k < 16; ++k) {
        int h = lane + 32 * k;
        tile[h * 9 + wid] = (tile[h * 9 + wid] - mu) * rs;
    }
    __syncthreads();
    for (int p = 0; p < 4; ++p) {
        int h = p * 128 + (tid >> 1);
        int part = (tid & 1) * 4;
        float wv = w[h], bv = bp[h];
        float4 o;
        o.x = tile[h * 9 + part + 0] * wv + bv;
        o.y = tile[h * 9 + part + 1] * wv + bv;
        o.z = tile[h * 9 + part + 2] * wv + bv;
        o.w = tile[h * 9 + part + 3] * wv + bv;
        size_t base = ((size_t)b * cH + h) * cL + l0 + part;
        *(float4*)&Z[base] = o;
        Zh[base + 0] = f2bf(o.x); Zh[base + 1] = f2bf(o.y);
        Zh[base + 2] = f2bf(o.z); Zh[base + 3] = f2bf(o.w);
    }
}

// ---------------- diagonal SSM recurrent scan + D*z + GELU -> bf16 Y --------------
__global__ __launch_bounds__(256) void scan_kernel(const float* __restrict__ Z,
                                                   const float4* __restrict__ coef,
                                                   const float* __restrict__ Dp,
                                                   ushort_t* __restrict__ Yh) {
    __shared__ float buf[256 * 33];
    int tid = threadIdx.x;
    size_t row0 = (size_t)blockIdx.x * 256;
    int h = (int)((row0 + tid) & (cH - 1));
    float dr[cN], di[cN], cr[cN], ci[cN], sr[cN], si[cN];
#pragma unroll
    for (int n = 0; n < cN; ++n) {
        float4 c = coef[h * cN + n];
        dr[n] = c.x; di[n] = c.y; cr[n] = c.z; ci[n] = c.w;
        sr[n] = 0.f; si[n] = 0.f;
    }
    float Dv = Dp[h];
    for (int chunk = 0; chunk < 32; ++chunk) {
        int l0 = chunk * 32;
        __syncthreads();
        for (int p = 0; p < 32; ++p) {
            int rl = p * 8 + (tid >> 5);
            int col = tid & 31;
            buf[rl * 33 + col] = Z[((row0 + rl) << 10) + l0 + col];
        }
        __syncthreads();
        float* my = &buf[tid * 33];
        for (int i = 0; i < 32; ++i) {
            float z = my[i];
            float yl = 0.f;
#pragma unroll
            for (int n = 0; n < cN; ++n) {
                float nsr = dr[n] * sr[n] - di[n] * si[n] + z;
                float nsi = dr[n] * si[n] + di[n] * sr[n];
                sr[n] = nsr; si[n] = nsi;
                yl += cr[n] * nsr - ci[n] * nsi;
            }
            my[i] = gelu_t(yl + Dv * z);
        }
        __syncthreads();
        for (int p = 0; p < 32; ++p) {
            int rl = p * 8 + (tid >> 5);
            int col = tid & 31;
            Yh[((row0 + rl) << 10) + l0 + col] = f2bf(buf[rl * 33 + col]);
        }
    }
}

// =========== WMMA bf16 GEMM machinery (async-LDS A staging, bf16 everywhere) ======
union AFrag { v16bf v; unsigned u[8]; };
constexpr int ASTR = 40; // padded LDS row stride (ushorts): 80B, 16B aligned
constexpr int BSTR = 34;

// stage A tile 128x32 bf16 via global_load_async_to_lds_b128 (2 chunks/thread)
__device__ __forceinline__ void stage_a_async(ushort_t* As, const ushort_t* A,
                                              int m0, int k0, int K, int tid) {
    unsigned base = (unsigned)(uintptr_t)As;
#pragma unroll
    for (int q = 0; q < 2; ++q) {
        int cc = tid * 2 + q;
        int row = cc >> 2, part = cc & 3;
        async_b128(base + (unsigned)(row * ASTR + part * 8) * 2,
                   &A[(size_t)(m0 + row) * K + k0 + part * 8]);
    }
}
// stage B tile 32x64 bf16 -> LDS transposed [n][k] (1x b128 load + 8 scatter stores)
__device__ __forceinline__ void stage_b(ushort_t* Bs, const ushort_t* Bp,
                                        int n0, int k0, int tid) {
    int k = tid >> 3, ns = (tid & 7) * 8;
    const uint4 v = *(const uint4*)&Bp[(size_t)(k0 + k) * cL + n0 + ns];
    unsigned vv[4] = {v.x, v.y, v.z, v.w};
#pragma unroll
    for (int q = 0; q < 4; ++q) {
        Bs[(ns + 2 * q + 0) * BSTR + k] = (ushort_t)(vv[q] & 0xffffu);
        Bs[(ns + 2 * q + 1) * BSTR + k] = (ushort_t)(vv[q] >> 16);
    }
}
__device__ __forceinline__ void load_a_frag(AFrag& a, const ushort_t* As, int mrow, int hi) {
#pragma unroll
    for (int v = 0; v < 8; ++v) {
        int kb = ((v >> 2) << 4) + (hi << 3) + ((v & 3) << 1);
        a.u[v] = *(const unsigned*)&As[mrow * ASTR + kb];
    }
}
__device__ __forceinline__ void load_b_frag(AFrag& b, const ushort_t* Bs, int ncol, int hi) {
#pragma unroll
    for (int v = 0; v < 8; ++v)
        b.u[v] = *(const unsigned*)&Bs[ncol * BSTR + ((v >> 2) << 4) + (hi << 3) + ((v & 3) << 1)];
}

// MODE 0: bf16 hdn = gelu(acc+bias) at m*L+n | MODE 1: f32 Out[m*L+n] += acc+bias
// MODE 2: f32 Out[n*V+m] = acc+bias (vocab head)
template <int MODE>
__global__ __launch_bounds__(256) void gemm_kernel(const ushort_t* __restrict__ A,
                                                   const ushort_t* __restrict__ Bact,
                                                   const float* __restrict__ bias,
                                                   void* __restrict__ Out,
                                                   int K, size_t strideB, size_t strideOut) {
    __shared__ __align__(16) ushort_t As[128 * ASTR];
    __shared__ __align__(16) ushort_t Bs[64 * BSTR];
    const ushort_t* Bp = Bact + (size_t)blockIdx.z * strideB;
    int m0 = blockIdx.y * 128, n0 = blockIdx.x * 64;
    int tid = threadIdx.x, wid = tid >> 5, lane = tid & 31;
    int waveM = (wid & 3) * 32, waveN = (wid >> 2) * 32;
    int hi = lane >> 4, ml = lane & 15;

    v8f acc[2][2];
    v8f zero = {};
    acc[0][0] = zero; acc[0][1] = zero; acc[1][0] = zero; acc[1][1] = zero;

    for (int k0 = 0; k0 < K; k0 += 32) {
        __syncthreads();
        stage_a_async(As, A, m0, k0, K, tid);
        stage_b(Bs, Bp, n0, k0, tid);
        if (k0 + 32 < K)
            __builtin_prefetch(&Bp[(size_t)(k0 + 32 + (tid >> 3)) * cL + n0], 0, 1);
        async_wait0();
        __syncthreads();
        AFrag a[2], bm[2];
#pragma unroll
        for (int f = 0; f < 2; ++f) {
            load_a_frag(a[f], As, waveM + f * 16 + ml, hi);
            load_b_frag(bm[f], Bs, waveN + f * 16 + ml, hi);
        }
#pragma unroll
        for (int i = 0; i < 2; ++i)
#pragma unroll
            for (int j = 0; j < 2; ++j)
                acc[i][j] = __builtin_amdgcn_wmma_f32_16x16x32_bf16(
                    false, a[i].v, false, bm[j].v, (short)0, acc[i][j], false, false);
    }

#pragma unroll
    for (int i = 0; i < 2; ++i)
#pragma unroll
        for (int j = 0; j < 2; ++j) {
            int nn = n0 + waveN + j * 16 + ml;
            v8f ac = acc[i][j];
#pragma unroll
            for (int r = 0; r < 8; ++r) {
                int mm = m0 + waveM + i * 16 + (hi << 3) + r;
                float val = ac[r] + bias[mm];
                if (MODE == 0) {
                    ((ushort_t*)Out + (size_t)blockIdx.z * strideOut)[(size_t)mm * cL + nn] =
                        f2bf(gelu_t(val));
                } else if (MODE == 1) {
                    ((float*)Out + (size_t)blockIdx.z * strideOut)[(size_t)mm * cL + nn] += val;
                } else {
                    ((float*)Out + (size_t)blockIdx.z * strideOut)[(size_t)nn * cV + mm] = val;
                }
            }
        }
}

// ---- WMMA GLU GEMM: X += (Wout[g]@Y + b[g]) * sigmoid(Wout[g+H]@Y + b[g+H]) ------
__global__ __launch_bounds__(256) void glu_kernel(const ushort_t* __restrict__ Wout, // bf16 [2H,H]
                                                  const float* __restrict__ bout,
                                                  const ushort_t* __restrict__ Yact, // bf16 (B,H,L)
                                                  float* __restrict__ X) {
    __shared__ __align__(16) ushort_t As1[128 * ASTR];
    __shared__ __align__(16) ushort_t As2[128 * ASTR];
    __shared__ __align__(16) ushort_t Bs[64 * BSTR];
    const ushort_t* Bp = Yact + (size_t)blockIdx.z * cH * cL;
    float* Xp = X + (size_t)blockIdx.z * cH * cL;
    int m0 = blockIdx.y * 128, n0 = blockIdx.x * 64;
    int tid = threadIdx.x, wid = tid >> 5, lane = tid & 31;
    int waveM = (wid & 3) * 32, waveN = (wid >> 2) * 32;
    int hi = lane >> 4, ml = lane & 15;

    v8f acc1[2][2], acc2[2][2];
    v8f zero = {};
#pragma unroll
    for (int i = 0; i < 2; ++i)
        for (int j = 0; j < 2; ++j) { acc1[i][j] = zero; acc2[i][j] = zero; }

    for (int k0 = 0; k0 < cH; k0 += 32) {
        __syncthreads();
        stage_a_async(As1, Wout, m0, k0, cH, tid);
        stage_a_async(As2, Wout, m0 + cH, k0, cH, tid);
        stage_b(Bs, Bp, n0, k0, tid);
        async_wait0();
        __syncthreads();
        AFrag a1[2], a2[2], bm[2];
#pragma unroll
        for (int f = 0; f < 2; ++f) {
            load_a_frag(a1[f], As1, waveM + f * 16 + ml, hi);
            load_a_frag(a2[f], As2, waveM + f * 16 + ml, hi);
            load_b_frag(bm[f], Bs, waveN + f * 16 + ml, hi);
        }
#pragma unroll
        for (int i = 0; i < 2; ++i)
#pragma unroll
            for (int j = 0; j < 2; ++j) {
                acc1[i][j] = __builtin_amdgcn_wmma_f32_16x16x32_bf16(
                    false, a1[i].v, false, bm[j].v, (short)0, acc1[i][j], false, false);
                acc2[i][j] = __builtin_amdgcn_wmma_f32_16x16x32_bf16(
                    false, a2[i].v, false, bm[j].v, (short)0, acc2[i][j], false, false);
            }
    }

#pragma unroll
    for (int i = 0; i < 2; ++i)
#pragma unroll
        for (int j = 0; j < 2; ++j) {
            int nn = n0 + waveN + j * 16 + ml;
#pragma unroll
            for (int r = 0; r < 8; ++r) {
                int mm = m0 + waveM + i * 16 + (hi << 3) + r;
                float v1 = acc1[i][j][r] + bout[mm];
                float v2 = acc2[i][j][r] + bout[mm + cH];
                Xp[(size_t)mm * cL + nn] += v1 * sigm(v2);
            }
        }
}

// ---------------- host orchestration ----------------
extern "C" void kernel_launch(void* const* d_in, const int* in_sizes, int n_in,
                              void* d_out, int out_size, void* d_ws, size_t ws_size,
                              hipStream_t stream) {
    (void)in_sizes; (void)n_in; (void)out_size; (void)ws_size;
    const int*   tokens    = (const int*)d_in[0];
    const float* emb       = (const float*)d_in[1];
    const float* s4_ln_w   = (const float*)d_in[2];
    const float* s4_ln_b   = (const float*)d_in[3];
    const float* s4_log_dt = (const float*)d_in[4];
    const float* s4_A_re   = (const float*)d_in[5];
    const float* s4_A_im   = (const float*)d_in[6];
    const float* s4_C_re   = (const float*)d_in[7];
    const float* s4_C_im   = (const float*)d_in[8];
    const float* s4_D      = (const float*)d_in[9];
    const float* s4_Wout   = (const float*)d_in[10];
    const float* s4_bout   = (const float*)d_in[11];
    const float* ff_ln_w   = (const float*)d_in[12];
    const float* ff_ln_b   = (const float*)d_in[13];
    const float* ff_W1     = (const float*)d_in[14];
    const float* ff_b1     = (const float*)d_in[15];
    const float* ff_W2     = (const float*)d_in[16];
    const float* ff_b2     = (const float*)d_in[17];
    const float* out_W     = (const float*)d_in[18];
    const float* out_b     = (const float*)d_in[19];

    // workspace layout
    float*    X    = (float*)d_ws;                       // (B,H,L) f32 residual
    float*    Zf   = X + NE;                             // (B,H,L) f32 LN out (scan input)
    ushort_t* Zh   = (ushort_t*)(Zf + NE);               // (B,H,L) bf16 LN out (GEMM B)
    ushort_t* ActB = Zh + NE;                            // bf16: Y / hdn chunk / X-bf16
    ushort_t* WoutB = ActB + NE;                         // bf16 weights
    constexpr size_t WOUT_N = (size_t)cNS4 * 2 * cH * cH;     // 4,194,304
    constexpr size_t W1_N   = (size_t)cLayers * cDFF * cH;    // 4,194,304
    constexpr size_t W2_N   = (size_t)cLayers * cH * cDFF;    // 4,194,304
    constexpr size_t OW_N   = (size_t)cV * cH;                // 65,536
    ushort_t* W1B = WoutB + WOUT_N;
    ushort_t* W2B = W1B + W1_N;
    ushort_t* OWB = W2B + W2_N;
    float4*   coef = (float4*)(OWB + OW_N);

    // one-time (per launch) bf16 weight conversion
    cvt_kernel<<<(unsigned)(WOUT_N / 256), 256, 0, stream>>>(s4_Wout, WoutB);
    cvt_kernel<<<(unsigned)(W1_N / 256), 256, 0, stream>>>(ff_W1, W1B);
    cvt_kernel<<<(unsigned)(W2_N / 256), 256, 0, stream>>>(ff_W2, W2B);
    cvt_kernel<<<(unsigned)(OW_N / 256), 256, 0, stream>>>(out_W, OWB);

    embed_kernel<<<(unsigned)(NE / 256), 256, 0, stream>>>(tokens, emb, X);

    for (int layer = 0; layer < cLayers; ++layer) {
        for (int s = 0; s < 2; ++s) {
            int j = 2 * layer + s;
            coef_kernel<<<(cH * cN) / 256, 256, 0, stream>>>(
                s4_log_dt + (size_t)j * cH,
                s4_A_re + (size_t)j * cNSSM * cN, s4_A_im + (size_t)j * cNSSM * cN,
                s4_C_re + (size_t)j * cH * cN,    s4_C_im + (size_t)j * cH * cN, coef);
            ln_kernel<<<dim3(cL / 8, cB), 256, 0, stream>>>(
                X, s4_ln_w + (size_t)j * cH, s4_ln_b + (size_t)j * cH, Zf, Zh);
            scan_kernel<<<(cB * cH) / 256, 256, 0, stream>>>(
                Zf, coef, s4_D + (size_t)j * cH, ActB);
            glu_kernel<<<dim3(cL / 64, cH / 128, cB), 256, 0, stream>>>(
                WoutB + (size_t)j * 2 * cH * cH, s4_bout + (size_t)j * 2 * cH, ActB, X);
        }
        // feed-forward block, batch-chunked so bf16 hdn reuses ActB
        ln_kernel<<<dim3(cL / 8, cB), 256, 0, stream>>>(
            X, ff_ln_w + (size_t)layer * cH, ff_ln_b + (size_t)layer * cH, Zf, Zh);
        for (int bc = 0; bc < 4; ++bc) {
            gemm_kernel<0><<<dim3(cL / 64, cDFF / 128, 16), 256, 0, stream>>>(
                W1B + (size_t)layer * cDFF * cH,
                Zh + (size_t)bc * 16 * cH * cL,
                ff_b1 + (size_t)layer * cDFF,
                ActB, cH, (size_t)cH * cL, (size_t)cDFF * cL);
            gemm_kernel<1><<<dim3(cL / 64, cH / 128, 16), 256, 0, stream>>>(
                W2B + (size_t)layer * cH * cDFF,
                ActB,
                ff_b2 + (size_t)layer * cH,
                X + (size_t)bc * 16 * cH * cL, cDFF, (size_t)cDFF * cL, (size_t)cH * cL);
        }
    }
    // vocab head: bf16 copy of X, then logits[b,l,v] = X[b,:,l].out_W[v,:] + out_b[v]
    cvt_kernel<<<(unsigned)(NE / 256), 256, 0, stream>>>(X, ActB);
    gemm_kernel<2><<<dim3(cL / 64, cV / 128, cB), 256, 0, stream>>>(
        OWB, ActB, out_b, d_out, cH, (size_t)cH * cL, (size_t)cL * cV);
}
